// MultiHeadAttention_61701500175237
// MI455X (gfx1250) — matrix operand order repaired and checked by reading, not compiled, not run on hardware
//
#include <hip/hip_runtime.h>

typedef unsigned short u16;
typedef __attribute__((ext_vector_type(16))) __bf16 v16bf;
typedef __attribute__((ext_vector_type(8)))  float  v8f;
typedef __attribute__((ext_vector_type(4)))  int    v4i;

union FragB { v16bf v; uint4 q[2]; };

#define DEV static __device__ __forceinline__

constexpr int kB  = 2;
constexpr int kS  = 2048;
constexpr int kDM = 1024;
constexpr int kH  = 16;
constexpr int kDK = 64;
constexpr int kM  = kB * kS; // 4096 rows of x / attn

// fp32 -> bf16, round-to-nearest-even
DEV u16 f2bf(float x) {
  unsigned u = __float_as_uint(x);
  u += 0x7FFFu + ((u >> 16) & 1u);
  return (u16)(u >> 16);
}

DEV v8f wmma_bf16(const FragB& a, const FragB& b, v8f c) {
  return __builtin_amdgcn_wmma_f32_16x16x32_bf16(false, a.v, false, b.v,
                                                 (short)0, c, false, false);
}

// ---- 16B global->LDS staging: async (ASYNCcnt-tracked) when available ----
#if __has_builtin(__builtin_amdgcn_global_load_async_to_lds_b128)
#define USE_ASYNC_LDS 1
#endif

DEV void stage16(const u16* g, u16* l) {
#ifdef USE_ASYNC_LDS
  __builtin_amdgcn_global_load_async_to_lds_b128(
      (__attribute__((address_space(1))) v4i*)g,
      (__attribute__((address_space(3))) v4i*)l, 0, 0);
#else
  *reinterpret_cast<uint4*>(l) = *reinterpret_cast<const uint4*>(g);
#endif
}

DEV void stage_join() {
#ifdef USE_ASYNC_LDS
#if __has_builtin(__builtin_amdgcn_s_wait_asynccnt)
  __builtin_amdgcn_s_wait_asynccnt(0);
#else
  asm volatile("s_wait_asynccnt 0x0" ::: "memory");
#endif
#endif
}

// ---------------------------------------------------------------- convert
__global__ void cvt_f32_bf16(const float* __restrict__ in, u16* __restrict__ out, int n) {
  int i = blockIdx.x * blockDim.x + threadIdx.x;
  if (i < n) out[i] = f2bf(in[i]);
}

// ------------------------------------------- GEMM: C[M,N] = A[M,K] * B[N,K]^T
// BM=128 BN=128 BK=32, 256 threads = 8 waves arranged 4(M) x 2(N);
// each wave computes a 32x64 tile = 2x4 wmma tiles (8 wmma / k-step).
__global__ __launch_bounds__(256) void gemm_nt(const u16* __restrict__ A,
                                               const u16* __restrict__ Bw,
                                               float* __restrict__ C,
                                               int M, int N, int K) {
  __shared__ __align__(16) u16 sA[128 * 32];
  __shared__ __align__(16) u16 sB[128 * 32];

  const int tid  = threadIdx.x;
  const int lane = tid & 31;
  const int w    = tid >> 5;
  const int g    = lane >> 4;   // lane group (hi/lo 16)
  const int ln   = lane & 15;
  const int wm   = w & 3;       // 4 M-groups of 32
  const int wn   = w >> 2;      // 2 N-groups of 64
  const int m0   = blockIdx.y * 128;
  const int n0   = blockIdx.x * 128;

  const v8f vzero = {0.f, 0.f, 0.f, 0.f, 0.f, 0.f, 0.f, 0.f};
  v8f acc[8];
#pragma unroll
  for (int i = 0; i < 8; ++i) acc[i] = vzero;

  const int srow = tid >> 1, sseg = (tid & 1) * 16; // 32B/thread per tile

  for (int k0 = 0; k0 < K; k0 += 32) {
    // stage A tile 128x32 and B tile 128x32
    stage16(&A[(size_t)(m0 + srow) * K + k0 + sseg],      &sA[srow * 32 + sseg]);
    stage16(&A[(size_t)(m0 + srow) * K + k0 + sseg + 8],  &sA[srow * 32 + sseg + 8]);
    stage16(&Bw[(size_t)(n0 + srow) * K + k0 + sseg],     &sB[srow * 32 + sseg]);
    stage16(&Bw[(size_t)(n0 + srow) * K + k0 + sseg + 8], &sB[srow * 32 + sseg + 8]);
    stage_join();
    __syncthreads();

    FragB fa[2], fb[4];
#pragma unroll
    for (int ms = 0; ms < 2; ++ms) { // A frag: lane=M row, halves K=(h&7)+g*8+(h>>3)*16
      int row = wm * 32 + ms * 16 + ln;
      fa[ms].q[0] = *reinterpret_cast<const uint4*>(&sA[row * 32 + g * 8]);
      fa[ms].q[1] = *reinterpret_cast<const uint4*>(&sA[row * 32 + 16 + g * 8]);
    }
#pragma unroll
    for (int ns = 0; ns < 4; ++ns) { // B frag: lane=N col, halves K=(h&7)+g*8+(h>>3)*16
      int row = wn * 64 + ns * 16 + ln;
      fb[ns].q[0] = *reinterpret_cast<const uint4*>(&sB[row * 32 + g * 8]);
      fb[ns].q[1] = *reinterpret_cast<const uint4*>(&sB[row * 32 + 16 + g * 8]);
    }
#pragma unroll
    for (int ms = 0; ms < 2; ++ms)
#pragma unroll
      for (int ns = 0; ns < 4; ++ns)
        acc[ms * 4 + ns] = wmma_bf16(fa[ms], fb[ns], acc[ms * 4 + ns]);
    __syncthreads();
  }

#pragma unroll
  for (int ms = 0; ms < 2; ++ms)
#pragma unroll
    for (int ns = 0; ns < 4; ++ns) {
      v8f o = acc[ms * 4 + ns];
#pragma unroll
      for (int r = 0; r < 8; ++r) { // C layout: M = r + g*8, N = ln
        int row = m0 + wm * 32 + ms * 16 + r + g * 8;
        int col = n0 + wn * 64 + ns * 16 + ln;
        C[(size_t)row * N + col] = o[r];
      }
    }
}

// ------------------------------------------- RoPE + permute to (B,H,S,64) bf16
__global__ void rope_permute(const float* __restrict__ in, u16* __restrict__ out,
                             float scale) {
  int idx = blockIdx.x * blockDim.x + threadIdx.x;
  if (idx >= kB * kH * kS) return;
  int s  = idx & (kS - 1);
  int bh = idx / kS;
  int h  = bh & (kH - 1);
  int b  = bh / kH;
  const float* src = in + ((size_t)(b * kS + s)) * kDM + h * kDK;
  u16* dst = out + ((size_t)bh * kS + s) * kDK;
#pragma unroll
  for (int i = 0; i < 32; ++i) {
    float x1 = src[2 * i], x2 = src[2 * i + 1];
    float fr = __powf(10000.0f, -(float)i * (1.0f / 32.0f));
    float ang = (float)s * fr;
    float sn, cs;
    __sincosf(ang, &sn, &cs);
    dst[2 * i]     = f2bf((x1 * cs - x2 * sn) * scale);
    dst[2 * i + 1] = f2bf((x1 * sn + x2 * cs) * scale);
  }
}

// ------------------------------------------- V permute to (B,H,64,S) bf16 (transposed)
__global__ void permute_v(const float* __restrict__ in, u16* __restrict__ out) {
  int idx = blockIdx.x * blockDim.x + threadIdx.x;
  if (idx >= kB * kH * kS) return;
  int s  = idx & (kS - 1);
  int bh = idx / kS;
  int h  = bh & (kH - 1);
  int b  = bh / kH;
  const float* src = in + ((size_t)(b * kS + s)) * kDM + h * kDK;
#pragma unroll
  for (int d = 0; d < kDK; ++d)
    out[((size_t)bh * kDK + d) * kS + s] = f2bf(src[d]);
}

// ------------------------------------------- causal flash attention
// grid (S/64, B*H); 128 threads = 4 waves, each wave owns 16 query rows.
// Scale 1/sqrt(64) pre-folded into Q.
__global__ __launch_bounds__(128) void flash_attn(const u16* __restrict__ Qb,
                                                  const u16* __restrict__ Kb,
                                                  const u16* __restrict__ Vt,
                                                  u16* __restrict__ attn) {
  __shared__ __align__(16) u16 sK[32 * 64];     // keys x d
  __shared__ __align__(16) u16 sV[64 * 32];     // d x keys (V^T)
  __shared__ __align__(16) u16 sP[4][16 * 32];  // per-wave P transpose scratch

  const int tid  = threadIdx.x;
  const int lane = tid & 31;
  const int w    = tid >> 5;
  const int g    = lane >> 4;
  const int ln   = lane & 15;
  const int qt   = blockIdx.x;
  const int bh   = blockIdx.y;
  const int q0   = qt * 64 + w * 16;

  // Q fragments: 16 rows x 64 d as two 16x32 A-frags
  FragB fq[2];
  {
    const u16* qrow = Qb + ((size_t)bh * kS + q0 + ln) * kDK;
#pragma unroll
    for (int c = 0; c < 2; ++c) {
      fq[c].q[0] = *reinterpret_cast<const uint4*>(qrow + c * 32 + g * 8);
      fq[c].q[1] = *reinterpret_cast<const uint4*>(qrow + c * 32 + 16 + g * 8);
    }
  }

  const v8f vzero = {0.f, 0.f, 0.f, 0.f, 0.f, 0.f, 0.f, 0.f};
  v8f accO[4];
  float mrow[8], lrow[8];
#pragma unroll
  for (int i = 0; i < 4; ++i) accO[i] = vzero;
#pragma unroll
  for (int r = 0; r < 8; ++r) { mrow[r] = -1e30f; lrow[r] = 0.f; }

  const int krow = tid >> 2, kseg = (tid & 3) * 16; // K stage: 32B/thread
  const int vrow = tid >> 1, vseg = (tid & 1) * 16; // V stage: 32B/thread

  const int nkt = 2 * qt + 2; // 32-key tiles covering the causal extent
  for (int kt = 0; kt < nkt; ++kt) {
    const int kb0 = kt * 32;
    // stage K tile 32 x 64 and V^T tile 64 x 32
    stage16(&Kb[((size_t)bh * kS + kb0 + krow) * kDK + kseg],     &sK[krow * 64 + kseg]);
    stage16(&Kb[((size_t)bh * kS + kb0 + krow) * kDK + kseg + 8], &sK[krow * 64 + kseg + 8]);
    stage16(&Vt[((size_t)bh * kDK + vrow) * kS + kb0 + vseg],     &sV[vrow * 32 + vseg]);
    stage16(&Vt[((size_t)bh * kDK + vrow) * kS + kb0 + vseg + 8], &sV[vrow * 32 + vseg + 8]);
    stage_join();
    __syncthreads();

    // S = Q K^T : 16 x 32 (two 16x16 tiles), contraction over d=64 (2 wmma each)
    v8f s[2];
#pragma unroll
    for (int ns = 0; ns < 2; ++ns) {
      s[ns] = vzero;
#pragma unroll
      for (int c = 0; c < 2; ++c) {
        FragB fk; // B frag: lane = key col, halves = d ((h&7)+g*8+(h>>3)*16) of chunk c
        fk.q[0] = *reinterpret_cast<const uint4*>(&sK[(ns * 16 + ln) * 64 + c * 32 + g * 8]);
        fk.q[1] = *reinterpret_cast<const uint4*>(&sK[(ns * 16 + ln) * 64 + c * 32 + 16 + g * 8]);
        s[ns] = wmma_bf16(fq[c], fk, s[ns]);
      }
    }

    // causal mask + online softmax (row stats shared by 16-lane groups)
#pragma unroll
    for (int r = 0; r < 8; ++r) {
      int q = q0 + r + g * 8;
      float s0 = (kb0 + ln <= q)      ? s[0][r] : -1e30f;
      float s1 = (kb0 + 16 + ln <= q) ? s[1][r] : -1e30f;
      float mx = fmaxf(s0, s1);
#pragma unroll
      for (int off = 1; off < 16; off <<= 1)
        mx = fmaxf(mx, __shfl_xor(mx, off, 16));
      float mnew  = fmaxf(mrow[r], mx);
      float alpha = __expf(mrow[r] - mnew);
      float p0 = __expf(s0 - mnew);
      float p1 = __expf(s1 - mnew);
      float rs = p0 + p1;
#pragma unroll
      for (int off = 1; off < 16; off <<= 1)
        rs += __shfl_xor(rs, off, 16);
      lrow[r] = lrow[r] * alpha + rs;
      mrow[r] = mnew;
#pragma unroll
      for (int ds = 0; ds < 4; ++ds) accO[ds][r] *= alpha;
      sP[w][(r + g * 8) * 32 + ln]      = f2bf(p0);
      sP[w][(r + g * 8) * 32 + 16 + ln] = f2bf(p1);
    }
    __syncthreads();

    // O += P * V : P as 16x32 A-frag, V^T gives contiguous B-frags per 16-d tile
    FragB fp;
    fp.q[0] = *reinterpret_cast<const uint4*>(&sP[w][ln * 32 + g * 8]);
    fp.q[1] = *reinterpret_cast<const uint4*>(&sP[w][ln * 32 + 16 + g * 8]);
#pragma unroll
    for (int ds = 0; ds < 4; ++ds) {
      FragB fv;
      fv.q[0] = *reinterpret_cast<const uint4*>(&sV[(ds * 16 + ln) * 32 + g * 8]);
      fv.q[1] = *reinterpret_cast<const uint4*>(&sV[(ds * 16 + ln) * 32 + 16 + g * 8]);
      accO[ds] = wmma_bf16(fp, fv, accO[ds]);
    }
    __syncthreads();
  }

  // normalize and write attn (B,S,D_MODEL) bf16 for the output projection GEMM
  const int b = bh >> 4, h = bh & 15;
#pragma unroll
  for (int r = 0; r < 8; ++r) {
    int q = q0 + r + g * 8;
    float inv = 1.0f / lrow[r];
    u16* orow = attn + ((size_t)(b * kS + q)) * kDM + h * kDK;
#pragma unroll
    for (int ds = 0; ds < 4; ++ds)
      orow[ds * 16 + ln] = f2bf(accO[ds][r] * inv);
  }
}

// ---------------------------------------------------------------- launch
extern "C" void kernel_launch(void* const* d_in, const int* in_sizes, int n_in,
                              void* d_out, int out_size, void* d_ws, size_t ws_size,
                              hipStream_t stream) {
  (void)in_sizes; (void)n_in; (void)out_size; (void)ws_size;
  const float* x  = (const float*)d_in[0];
  const float* Wq = (const float*)d_in[1];
  const float* Wk = (const float*)d_in[2];
  const float* Wv = (const float*)d_in[3];
  const float* Wo = (const float*)d_in[4];
  float* out = (float*)d_out;

  char* ws = (char*)d_ws;
  size_t off = 0;
  auto take = [&](size_t bytes) { char* p = ws + off; off += bytes; return p; };
  u16*   xb  = (u16*)take((size_t)kM * kDM * 2);   // x bf16; reused later as attn
  u16*   Wqb = (u16*)take((size_t)kDM * kDM * 2);
  u16*   Wkb = (u16*)take((size_t)kDM * kDM * 2);
  u16*   Wvb = (u16*)take((size_t)kDM * kDM * 2);
  u16*   Wob = (u16*)take((size_t)kDM * kDM * 2);
  float* Qf  = (float*)take((size_t)kM * kDM * 4);
  float* Kf  = (float*)take((size_t)kM * kDM * 4);
  float* Vf  = (float*)take((size_t)kM * kDM * 4);
  u16*   Qb  = (u16*)take((size_t)kM * kDM * 2);   // (B,H,S,64)
  u16*   Kbf = (u16*)take((size_t)kM * kDM * 2);   // (B,H,S,64)
  u16*   Vtb = (u16*)take((size_t)kM * kDM * 2);   // (B,H,64,S)
  u16*   attnb = xb;                               // alias: xb dead after QKV GEMMs

  const int nX = kM * kDM, nW = kDM * kDM;
  cvt_f32_bf16<<<(nX + 255) / 256, 256, 0, stream>>>(x,  xb,  nX);
  cvt_f32_bf16<<<(nW + 255) / 256, 256, 0, stream>>>(Wq, Wqb, nW);
  cvt_f32_bf16<<<(nW + 255) / 256, 256, 0, stream>>>(Wk, Wkb, nW);
  cvt_f32_bf16<<<(nW + 255) / 256, 256, 0, stream>>>(Wv, Wvb, nW);
  cvt_f32_bf16<<<(nW + 255) / 256, 256, 0, stream>>>(Wo, Wob, nW);

  dim3 gGemm(kDM / 128, kM / 128);
  gemm_nt<<<gGemm, 256, 0, stream>>>(xb, Wqb, Qf, kM, kDM, kDM);
  gemm_nt<<<gGemm, 256, 0, stream>>>(xb, Wkb, Kf, kM, kDM, kDM);
  gemm_nt<<<gGemm, 256, 0, stream>>>(xb, Wvb, Vf, kM, kDM, kDM);

  const int nRows = kB * kH * kS;
  rope_permute<<<(nRows + 255) / 256, 256, 0, stream>>>(Qf, Qb, 0.125f); // fold 1/sqrt(dk)
  rope_permute<<<(nRows + 255) / 256, 256, 0, stream>>>(Kf, Kbf, 1.0f);
  permute_v  <<<(nRows + 255) / 256, 256, 0, stream>>>(Vf, Vtb);

  flash_attn<<<dim3(kS / 64, kB * kH), 128, 0, stream>>>(Qb, Kbf, Vtb, attnb);

  gemm_nt<<<gGemm, 256, 0, stream>>>(attnb, Wob, out, kM, kDM, kDM);
}
